// SiReN_42365557408249
// MI455X (gfx1250) — matrix operand order repaired and checked by reading, not compiled, not run on hardware
//
#include <hip/hip_runtime.h>
#include <hip/hip_bf16.h>
#include <math.h>

#define DIM 64

typedef __attribute__((ext_vector_type(16))) _Float16 v16h;
typedef __attribute__((ext_vector_type(8)))  float    v8f;

// ---------------------------------------------------------------- helpers

__device__ __forceinline__ v8f wmma_f16(v16h a, v16h b, v8f c) {
    // D = A(16x32 f16) * B(32x16 f16) + C(16x16 f32)
    return __builtin_amdgcn_wmma_f32_16x16x32_f16(
        /*neg_a=*/false, a, /*neg_b=*/false, b,
        /*c_mod=*/(short)0, c, /*reuse_a=*/false, /*reuse_b=*/false);
}

// A-fragment (16x32, f16) from an f32 row-major [*, 64] matrix.
// Layout (ISA 7.12.2): lane l: m=l%16, h=l/16; VGPR0-3 hold K=8h..8h+7,
// VGPR4-7 hold K=16+8h..16+8h+7.
__device__ __forceinline__ v16h load_a_f32(const float* X, int rowBase, int kb, int lane) {
    const int m = lane & 15, h = lane >> 4;
    const float* p = X + (rowBase + m) * DIM + kb;
    v16h a;
#pragma unroll
    for (int e = 0; e < 8; ++e) a[e] = (_Float16)p[8 * h + e];
#pragma unroll
    for (int e = 0; e < 8; ++e) a[8 + e] = (_Float16)p[16 + 8 * h + e];
    return a;
}

// Same A-fragment but from an f16 row-major matrix (LDS hidden tile).
__device__ __forceinline__ v16h load_a_f16(const _Float16* X, int rowBase, int kb, int lane) {
    const int m = lane & 15, h = lane >> 4;
    const _Float16* p = X + (rowBase + m) * DIM + kb;
    v16h a;
#pragma unroll
    for (int e = 0; e < 8; ++e) a[e] = p[8 * h + e];
#pragma unroll
    for (int e = 0; e < 8; ++e) a[8 + e] = p[16 + 8 * h + e];
    return a;
}

// B-fragment (32x16, f16) for out = X @ W^T, i.e. B[k][n] = W[n][k].
// B is loaded column-major: lane l: n=l%16, h=l/16; elements 0-7 hold K=8h..8h+7, 8-15 hold K=16+8h..16+8h+7.
__device__ __forceinline__ v16h load_b_wT(const _Float16* W, int nb, int kb, int lane) {
    const int n = lane & 15, h = lane >> 4;
    const _Float16* p = W + (nb + n) * DIM + kb;
    v16h b;
#pragma unroll
    for (int e = 0; e < 8; ++e) b[e] = p[8 * h + e];
#pragma unroll
    for (int e = 0; e < 8; ++e) b[8 + e] = p[16 + 8 * h + e];
    return b;
}

__device__ __forceinline__ float wave_sum(float v) {
#pragma unroll
    for (int m = 16; m >= 1; m >>= 1) v += __shfl_xor(v, m, 32);
    return v;
}

__device__ __forceinline__ void atomAddF(float* p, float v) {
    __hip_atomic_fetch_add(p, v, __ATOMIC_RELAXED, __HIP_MEMORY_SCOPE_AGENT);
}

// ---------------------------------------------------------------- kernels

__global__ void k_fill(float* p, long n) {
    long i  = (long)blockIdx.x * blockDim.x + threadIdx.x;
    long st = (long)gridDim.x * blockDim.x;
    for (; i < n; i += st) p[i] = 0.f;
}

__global__ void k_deg(const int* dst, int nE, float* deg) {
    int i  = blockIdx.x * blockDim.x + threadIdx.x;
    int st = gridDim.x * blockDim.x;
    for (; i < nE; i += st) atomAddF(&deg[dst[i]], 1.0f);
}

__global__ void k_dinv(float* deg, int N) {
    int i  = blockIdx.x * blockDim.x + threadIdx.x;
    int st = gridDim.x * blockDim.x;
    for (; i < N; i += st) {
        float d = deg[i];
        deg[i] = (d > 0.f) ? rsqrtf(fmaxf(d, 1.f)) : 0.f;
    }
}

// one wave per edge: out[dst] += dinv[src]*dinv[dst] * x[src]  (64 floats)
__global__ void k_prop(const float* X, const int* src, const int* dst,
                       const float* dinv, float* Y, int nE) {
    const int lane = threadIdx.x & 31;
    int wid = blockIdx.x * (blockDim.x >> 5) + (threadIdx.x >> 5);
    int nw  = gridDim.x * (blockDim.x >> 5);
    for (int e = wid; e < nE; e += nw) {
        int s = src[e], d = dst[e];                  // uniform -> scalar loads
        float nm = dinv[s] * dinv[d];
        const float* xr = X + s * DIM;
        float* yr = Y + d * DIM;
        atomAddF(&yr[lane],      nm * xr[lane]);
        atomAddF(&yr[lane + 32], nm * xr[lane + 32]);
    }
}

// b1 <- (E + b1 + b2) / 3   (in place on b1)
__global__ void k_zp(const float* E, float* b1, const float* b2, long n) {
    long i  = (long)blockIdx.x * blockDim.x + threadIdx.x;
    long st = (long)gridDim.x * blockDim.x;
    for (; i < n; i += st) b1[i] = (E[i] + b1[i] + b2[i]) * (1.f / 3.f);
}

// fused 2-layer MLP: Y = relu(relu(X@W0^T + b0)@W1^T + b1), via WMMA f16->f32
__global__ void __launch_bounds__(256) k_mlp2(const float* X, const float* W0, const float* B0,
                                              const float* W1, const float* B1,
                                              float* Y, int ntiles) {
    __shared__ _Float16 W0l[DIM * DIM];
    __shared__ _Float16 W1l[DIM * DIM];
    __shared__ float b0l[DIM], b1l[DIM];
    __shared__ _Float16 hl[8][16 * DIM];   // per-wave hidden tile
    for (int i = threadIdx.x; i < DIM * DIM; i += blockDim.x) {
        W0l[i] = (_Float16)W0[i];
        W1l[i] = (_Float16)W1[i];
    }
    if (threadIdx.x < DIM) { b0l[threadIdx.x] = B0[threadIdx.x]; b1l[threadIdx.x] = B1[threadIdx.x]; }
    __syncthreads();

    const int lane  = threadIdx.x & 31;
    const int wslot = threadIdx.x >> 5;
    _Float16* ht = hl[wslot];
    const int n0 = lane & 15;
    const int mh = (lane >> 4) * 8;
    int wid = blockIdx.x * (blockDim.x >> 5) + wslot;
    int nw  = gridDim.x * (blockDim.x >> 5);
    const v8f zero = {0.f, 0.f, 0.f, 0.f, 0.f, 0.f, 0.f, 0.f};

    for (int tile = wid; tile < ntiles; tile += nw) {
        const int rowBase = tile * 16;
        v8f c[4] = {zero, zero, zero, zero};
#pragma unroll
        for (int kb = 0; kb < DIM; kb += 32) {
            v16h a = load_a_f32(X, rowBase, kb, lane);
#pragma unroll
            for (int t = 0; t < 4; ++t)
                c[t] = wmma_f16(a, load_b_wT(W0l, t * 16, kb, lane), c[t]);
        }
        // bias + relu, hidden tile to LDS as f16 (A-matrix for layer 2)
#pragma unroll
        for (int r = 0; r < 8; ++r)
#pragma unroll
            for (int t = 0; t < 4; ++t) {
                float v = c[t][r] + b0l[t * 16 + n0];
                ht[(mh + r) * DIM + t * 16 + n0] = (_Float16)(v > 0.f ? v : 0.f);
            }
        v8f d[4] = {zero, zero, zero, zero};
#pragma unroll
        for (int kb = 0; kb < DIM; kb += 32) {
            v16h a = load_a_f16(ht, 0, kb, lane);
#pragma unroll
            for (int t = 0; t < 4; ++t)
                d[t] = wmma_f16(a, load_b_wT(W1l, t * 16, kb, lane), d[t]);
        }
#pragma unroll
        for (int r = 0; r < 8; ++r) {
            float* yr = Y + (rowBase + mh + r) * DIM;
#pragma unroll
            for (int t = 0; t < 4; ++t) {
                float v = d[t][r] + b1l[t * 16 + n0];
                yr[t * 16 + n0] = v > 0.f ? v : 0.f;
            }
        }
    }
}

// per 16-row tile: s[r] (r=0..7) = reduced attention score for row (8h + r)
__device__ __forceinline__ void attn_scores8(const float* X, int rowBase, int lane,
                                             const _Float16* Wl, const float* bl,
                                             const float* ql, float s[8]) {
    const v8f zero = {0.f, 0.f, 0.f, 0.f, 0.f, 0.f, 0.f, 0.f};
    v8f c[4] = {zero, zero, zero, zero};
#pragma unroll
    for (int kb = 0; kb < DIM; kb += 32) {
        v16h a = load_a_f32(X, rowBase, kb, lane);
#pragma unroll
        for (int t = 0; t < 4; ++t)
            c[t] = wmma_f16(a, load_b_wT(Wl, t * 16, kb, lane), c[t]);
    }
    const int n0 = lane & 15;
#pragma unroll
    for (int r = 0; r < 8; ++r) s[r] = 0.f;
#pragma unroll
    for (int t = 0; t < 4; ++t) {
        float qv = ql[t * 16 + n0];
        float bv = bl[t * 16 + n0];
#pragma unroll
        for (int r = 0; r < 8; ++r) s[r] += tanhf(c[t][r] + bv) * qv;
    }
    // reduce across the 16 columns held by each lane-half
#pragma unroll
    for (int r = 0; r < 8; ++r) {
        float v = s[r];
#pragma unroll
        for (int m = 8; m >= 1; m >>= 1) v += __shfl_xor(v, m, 16);
        s[r] = v;
    }
}

// attention fusion: Z = softmax([w_p,w_n]) blend of z_p, z_n (Z may alias Zp)
__global__ void __launch_bounds__(256) k_attn(const float* Zp, const float* Zn,
                                              const float* W, const float* bias,
                                              const float* q, float* Z, int ntiles) {
    __shared__ _Float16 Wl[DIM * DIM];
    __shared__ float bl[DIM], ql[DIM];
    for (int i = threadIdx.x; i < DIM * DIM; i += blockDim.x) Wl[i] = (_Float16)W[i];
    if (threadIdx.x < DIM) { bl[threadIdx.x] = bias[threadIdx.x]; ql[threadIdx.x] = q[threadIdx.x]; }
    __syncthreads();

    const int lane = threadIdx.x & 31;
    const int h = lane >> 4;
    int wid = blockIdx.x * (blockDim.x >> 5) + (threadIdx.x >> 5);
    int nw  = gridDim.x * (blockDim.x >> 5);

    for (int tile = wid; tile < ntiles; tile += nw) {
        const int rowBase = tile * 16;
        float sp[8], sn[8];
        attn_scores8(Zp, rowBase, lane, Wl, bl, ql, sp);
        attn_scores8(Zn, rowBase, lane, Wl, bl, ql, sn);
        float wp[16], wn[16];
#pragma unroll
        for (int r = 0; r < 8; ++r) {        // exchange halves: get all 16 rows
            float op = __shfl_xor(sp[r], 16, 32);
            float on = __shfl_xor(sn[r], 16, 32);
            wp[r]     = h ? op    : sp[r];
            wp[r + 8] = h ? sp[r] : op;
            wn[r]     = h ? on    : sn[r];
            wn[r + 8] = h ? sn[r] : on;
        }
#pragma unroll
        for (int r = 0; r < 16; ++r) {
            float mx = fmaxf(wp[r], wn[r]);
            float ep = __expf(wp[r] - mx), en = __expf(wn[r] - mx);
            float a0 = ep / (ep + en);
            int row = rowBase + r;
            float z0 = a0 * Zp[row * DIM + lane]      + (1.f - a0) * Zn[row * DIM + lane];
            float z1 = a0 * Zp[row * DIM + 32 + lane] + (1.f - a0) * Zn[row * DIM + 32 + lane];
            Z[row * DIM + lane]      = z0;
            Z[row * DIM + 32 + lane] = z1;
        }
    }
}

// one wave per batch item: pos/neg dots, sBPR log-sigmoid, L2 reg
__global__ void k_loss(const float* Z, const int* u, const int* v, const int* n,
                       const float* w, float* out, int B, int K) {
    const int lane = threadIdx.x & 31;
    int wid = blockIdx.x * (blockDim.x >> 5) + (threadIdx.x >> 5);
    int nw  = gridDim.x * (blockDim.x >> 5);
    float acc = 0.f;
    for (int b = wid; b < B; b += nw) {
        const float* ur = Z + u[b] * DIM;
        const float* vr = Z + v[b] * DIM;
        float u0 = ur[lane], u1 = ur[lane + 32];
        float v0 = vr[lane], v1 = vr[lane + 32];
        float pos = wave_sum(u0 * v0 + u1 * v1);
        float sq  = u0 * u0 + u1 * u1 + v0 * v0 + v1 * v1;
        float wb  = w[b];
        float sgn = (wb > 0.f) ? 1.f : ((wb < 0.f) ? -1.f : 0.f);
        float sp  = sgn * pos;
        float lsum = 0.f;
        const int* nb = n + b * K;
        for (int k = 0; k < K; ++k) {
            const float* nr = Z + nb[k] * DIM;
            float n0 = nr[lane], n1 = nr[lane + 32];
            sq += n0 * n0 + n1 * n1;
            float neg = wave_sum(u0 * n0 + u1 * n1);
            float x = sp - neg;
            lsum += fminf(x, 0.f) - log1pf(__expf(-fabsf(x)));   // log_sigmoid(x)
        }
        float sqs = wave_sum(sq);
        acc += -lsum + 1e-4f * sqs;
    }
    if (lane == 0) atomAddF(out, acc);
}

// ---------------------------------------------------------------- launch

extern "C" void kernel_launch(void* const* d_in, const int* in_sizes, int n_in,
                              void* d_out, int out_size, void* d_ws, size_t ws_size,
                              hipStream_t stream) {
    const float* E    = (const float*)d_in[0];
    const float* E2   = (const float*)d_in[1];
    const float* Wm   = (const float*)d_in[2];   // [2,64,64]
    const float* bm   = (const float*)d_in[3];   // [2,64]
    const float* aW   = (const float*)d_in[4];
    const float* ab   = (const float*)d_in[5];
    const float* qw   = (const float*)d_in[6];
    const int*   eidx = (const int*)d_in[7];     // [2, 2*N_INTER]
    const int*   u    = (const int*)d_in[8];
    const int*   v    = (const int*)d_in[9];
    const int*   nn   = (const int*)d_in[10];
    const float* w    = (const float*)d_in[11];

    const int  N  = in_sizes[0] / DIM;   // 150000 nodes
    const int  nE = in_sizes[7] / 2;     // 3.2M directed edges
    const int  B  = in_sizes[8];         // 16384
    const int  K  = in_sizes[10] / B;    // 40
    const long nd = (long)N * DIM;

    float* ws   = (float*)d_ws;
    float* dinv = ws;                         // [N]  (deg then dinv, in place)
    float* buf1 = ws + ((N + 255) & ~255);    // [N*64]: b1 -> z_p -> Z
    float* buf2 = buf1 + nd;                  // [N*64]: b2 -> z_n
    float* out  = (float*)d_out;
    const int* src = eidx;
    const int* dst = eidx + nE;

    // zero accumulators + output (graph-replay safe: done every launch)
    k_fill<<<1024, 256, 0, stream>>>(dinv, N);
    k_fill<<<4096, 256, 0, stream>>>(buf1, nd);
    k_fill<<<4096, 256, 0, stream>>>(buf2, nd);
    k_fill<<<1, 32, 0, stream>>>(out, 1);

    // degree -> dinv
    k_deg<<<2048, 256, 0, stream>>>(dst, nE, dinv);
    k_dinv<<<1024, 256, 0, stream>>>(dinv, N);

    // two LightGCN propagations (L2-resident: 38.4MB per node matrix)
    k_prop<<<4096, 256, 0, stream>>>(E,    src, dst, dinv, buf1, nE);
    k_prop<<<4096, 256, 0, stream>>>(buf1, src, dst, dinv, buf2, nE);

    // z_p = (E + b1 + b2)/3  (into buf1)
    k_zp<<<4096, 256, 0, stream>>>(E, buf1, buf2, nd);

    // z_n = MLP(E2)  (into buf2; fused 2 layers, WMMA)
    const int ntiles = N / 16;               // 150000 % 16 == 0
    const int gb = (ntiles + 7) / 8;
    k_mlp2<<<gb, 256, 0, stream>>>(E2, Wm, bm, Wm + DIM * DIM, bm + DIM, buf2, ntiles);

    // attention fusion -> Z (in place into buf1)
    k_attn<<<gb, 256, 0, stream>>>(buf1, buf2, aW, ab, qw, buf1, ntiles);

    // sBPR loss + regularization -> scalar
    k_loss<<<(B + 7) / 8, 256, 0, stream>>>(buf1, u, v, nn, w, out, B, K);
}
